// GPSNetNode_39402029973518
// MI455X (gfx1250) — matrix operand order repaired and checked by reading, not compiled, not run on hardware
//
#include <hip/hip_runtime.h>
#include <hip/hip_bf16.h>

#define NN 100000
#define EE 1600000
#define HH 128
#define CC 40
#define LL 4
#define BN_EPS 1e-5f

typedef __bf16 bf16_t;
typedef __attribute__((ext_vector_type(16))) __bf16 v16bf;
typedef __attribute__((ext_vector_type(8)))  __bf16 v8bf;
typedef __attribute__((ext_vector_type(8)))  float  v8f;
typedef __attribute__((ext_vector_type(4)))  float  v4f;

// ---------- bf16 helpers (bit-exact RNE) -------------------------------------
__device__ __forceinline__ bf16_t f2bf(float f) {
  unsigned u = __float_as_uint(f);
  unsigned r = (u + 0x7FFFu + ((u >> 16) & 1u)) >> 16;
  unsigned short s = (unsigned short)r;
  return *reinterpret_cast<bf16_t*>(&s);
}
__device__ __forceinline__ float bf2f(bf16_t b) {
  unsigned short s = *reinterpret_cast<unsigned short*>(&b);
  return __uint_as_float(((unsigned)s) << 16);
}

// ---------- WMMA fragment loaders --------------------------------------------
// A: 16x32 bf16 tile from row-major [*, 128] activations (global).
__device__ __forceinline__ v16bf ld_a_frag(const bf16_t* __restrict__ A,
                                           int m0, int k0, int lane) {
  const int row = m0 + (lane & 15);
  const int kb  = k0 + ((lane >> 4) << 3);
  const bf16_t* p = A + row * HH + kb;
  union { v16bf v; v8bf h[2]; } u;
  u.h[0] = *(const v8bf*)(p);
  u.h[1] = *(const v8bf*)(p + 16);
  return u.v;
}
// B: 32x16 bf16 tile from PRE-TRANSPOSED weights in LDS (row-major [N][128]).
__device__ __forceinline__ v16bf ld_b_frag_lds(const bf16_t* WT,
                                               int n0, int k0, int lane) {
  const int ncol = n0 + (lane & 15);
  const int kb   = k0 + ((lane >> 4) << 3);
  const bf16_t* p = WT + ncol * HH + kb;
  union { v16bf v; v8bf h[2]; } u;
  u.h[0] = *(const v8bf*)(p);
  u.h[1] = *(const v8bf*)(p + 16);
  return u.v;
}

extern __shared__ __align__(16) char gps_smem[];

// 8-tile pass over one K-step with a 2-deep B-fragment register pipeline:
// ds_load for tile t+2 is in flight while WMMA for tile t executes.
__device__ __forceinline__ void wmma_row_pass(const bf16_t* sW, v16bf a, int k0,
                                              int lane, v8f acc[8]) {
  v16bf b0 = ld_b_frag_lds(sW, 0, k0, lane);
  v16bf b1 = ld_b_frag_lds(sW, 16, k0, lane);
#pragma unroll
  for (int t = 0; t < 8; ++t) {
    v16bf bn = b1;
    if (t + 2 < 8) bn = ld_b_frag_lds(sW, (t + 2) * 16, k0, lane);
    acc[t] = __builtin_amdgcn_wmma_f32_16x16x32_bf16(
        false, a, false, b0, (short)0, acc[t], false, false);
    b0 = b1;
    b1 = bn;
  }
}

// ---------- small utility kernels --------------------------------------------
__global__ void gps_zero_f32(float* __restrict__ p, int n) {
  int i = blockIdx.x * blockDim.x + threadIdx.x;
  if (i < n) p[i] = 0.f;
}
__global__ void gps_zero_f32x4(v4f* __restrict__ p, int n4) {
  int i = blockIdx.x * blockDim.x + threadIdx.x;
  if (i < n4) p[i] = (v4f){0.f, 0.f, 0.f, 0.f};
}
__global__ void gps_cvt_bf16(const float* __restrict__ src, bf16_t* __restrict__ dst, int n) {
  int i = blockIdx.x * blockDim.x + threadIdx.x;
  if (i < n) dst[i] = f2bf(src[i]);
}
// WT[n*K + k] = W[k*Nc + n]  (zero-pad columns n >= Nc up to NcPad)
__global__ void gps_transpose_w(const float* __restrict__ W, bf16_t* __restrict__ WT,
                                int K, int Nc, int NcPad) {
  int i = blockIdx.x * blockDim.x + threadIdx.x;
  if (i >= NcPad * K) return;
  int n = i / K, k = i - n * K;
  WT[i] = (n < Nc) ? f2bf(W[k * Nc + n]) : f2bf(0.f);
}
__global__ void gps_degree(const int* __restrict__ dst, float* __restrict__ deg, int E) {
  int e = blockIdx.x * blockDim.x + threadIdx.x;
  if (e < E) atomicAdd(&deg[dst[e]], 1.0f);
}
__global__ void gps_invdeg(float* __restrict__ deg, int n) {
  int i = blockIdx.x * blockDim.x + threadIdx.x;
  if (i < n) deg[i] = 1.0f / fmaxf(deg[i], 1.0f);
}

// ---------- edge scatter: s[dst] += h[src] (f32 atomics in L2) ---------------
__global__ void gps_scatter(const bf16_t* __restrict__ h, const int* __restrict__ src,
                            const int* __restrict__ dst, float* __restrict__ s, int E) {
  const int t  = threadIdx.x;
  const int e0 = blockIdx.x * 32;
  for (int j = 0; j < 32; ++j) {
    int e = e0 + j;
    if (e >= E) return;
    if ((j & 3) == 0 && (e + 4) < E) {
      int sp = src[e + 4];
      __builtin_prefetch(h + sp * HH, 0, 0);   // global_prefetch_b8
    }
    int si = src[e];
    int di = dst[e];
    float v = bf2f(h[si * HH + t]);
    atomicAdd(&s[di * HH + t], v);
  }
}
__global__ void gps_mean_bf16(const float* __restrict__ s, const float* __restrict__ inv,
                              bf16_t* __restrict__ agg, int n) {
  int i = blockIdx.x * blockDim.x + threadIdx.x;
  if (i < n) agg[i] = f2bf(s[i] * inv[i >> 7]);
}

// ---------- fused (dual) GEMM: out = A1@W1 [+ A2@W2] + bias ------------------
// [N,128] x [128,128]. 8 waves/block (128 rows). Weights staged in LDS.
// Dynamic LDS: 32KB (single) or 64KB (dual).
__global__ void gps_gemm128(const bf16_t* __restrict__ A1, const bf16_t* __restrict__ W1T,
                            const bf16_t* __restrict__ A2, const bf16_t* __restrict__ W2T,
                            const float* __restrict__ bias, float* __restrict__ out, int N) {
  bf16_t* sW1 = (bf16_t*)gps_smem;
  bf16_t* sW2 = sW1 + HH * HH;

  { // cooperative staging: 2048 x 16B per matrix, 256 threads
    const uint4* g1 = (const uint4*)W1T;
    uint4* s1 = (uint4*)sW1;
    for (int i = threadIdx.x; i < (HH * HH) / 8; i += blockDim.x) s1[i] = g1[i];
    if (A2 != nullptr) {
      const uint4* g2 = (const uint4*)W2T;
      uint4* s2 = (uint4*)sW2;
      for (int i = threadIdx.x; i < (HH * HH) / 8; i += blockDim.x) s2[i] = g2[i];
    }
  }
  __syncthreads();

  const int lane = threadIdx.x & 31;
  const int wid  = blockIdx.x * (blockDim.x >> 5) + (threadIdx.x >> 5);
  const int m0   = wid * 16;
  if (m0 >= N) return;

  v8f acc[8] = {};

#pragma unroll
  for (int k0 = 0; k0 < HH; k0 += 32) {
    v16bf a = ld_a_frag(A1, m0, k0, lane);
    wmma_row_pass(sW1, a, k0, lane, acc);
  }
  if (A2 != nullptr) {
#pragma unroll
    for (int k0 = 0; k0 < HH; k0 += 32) {
      v16bf a = ld_a_frag(A2, m0, k0, lane);
      wmma_row_pass(sW2, a, k0, lane, acc);
    }
  }

  const int col_lo = lane & 15;
  const int rbase  = m0 + ((lane >> 4) << 3);
#pragma unroll
  for (int t = 0; t < 8; ++t) {
    float bv = bias[t * 16 + col_lo];
#pragma unroll
    for (int r = 0; r < 8; ++r)
      out[(rbase + r) * HH + t * 16 + col_lo] = acc[t][r] + bv;
  }
}

// ---------- output GEMM: [N,128] x [128,40] (padded to 48 cols) --------------
// Dynamic LDS: 12KB for staged WoT.
__global__ void gps_gemm_out(const bf16_t* __restrict__ A, const bf16_t* __restrict__ WoT,
                             const float* __restrict__ bias, float* __restrict__ out, int N) {
  bf16_t* sWo = (bf16_t*)gps_smem;
  {
    const uint4* g = (const uint4*)WoT;
    uint4* sm = (uint4*)sWo;
    for (int i = threadIdx.x; i < (48 * HH) / 8; i += blockDim.x) sm[i] = g[i];
  }
  __syncthreads();

  const int lane = threadIdx.x & 31;
  const int wid  = blockIdx.x * (blockDim.x >> 5) + (threadIdx.x >> 5);
  const int m0   = wid * 16;
  if (m0 >= N) return;

  v8f acc[3] = {};
#pragma unroll
  for (int k0 = 0; k0 < HH; k0 += 32) {
    v16bf a = ld_a_frag(A, m0, k0, lane);
    v16bf b0 = ld_b_frag_lds(sWo, 0, k0, lane);
    v16bf b1 = ld_b_frag_lds(sWo, 16, k0, lane);
#pragma unroll
    for (int t = 0; t < 3; ++t) {
      v16bf bn = b1;
      if (t + 2 < 3) bn = ld_b_frag_lds(sWo, (t + 2) * 16, k0, lane);
      acc[t] = __builtin_amdgcn_wmma_f32_16x16x32_bf16(
          false, a, false, b0, (short)0, acc[t], false, false);
      b0 = b1;
      b1 = bn;
    }
  }
  const int col_lo = lane & 15;
  const int rbase  = m0 + ((lane >> 4) << 3);
#pragma unroll
  for (int t = 0; t < 3; ++t) {
    int col = t * 16 + col_lo;
    if (col < CC) {
      float bv = bias[col];
#pragma unroll
      for (int r = 0; r < 8; ++r)
        out[(rbase + r) * CC + col] = acc[t][r] + bv;
    }
  }
}

// ---------- BatchNorm: column stats (LDS combine) + finalize + apply ---------
__global__ void gps_bnstats(const float* __restrict__ y, float* __restrict__ cs,
                            float* __restrict__ cq, int N) {
  __shared__ float ls[256];
  __shared__ float lq[256];
  const int t    = threadIdx.x;       // 256 threads: 2 per column
  const int c    = t & (HH - 1);
  const int half = t >> 7;
  const int r0   = blockIdx.x * 512;
  const int rend = (r0 + 512 < N) ? (r0 + 512) : N;
  float s = 0.f, q = 0.f;
  for (int r = r0 + half; r < rend; r += 2) {
    float v = y[r * HH + c];
    s += v; q += v * v;
  }
  ls[t] = s; lq[t] = q;
  __syncthreads();
  if (half == 0) {
    s += ls[t + 128];
    q += lq[t + 128];
    atomicAdd(&cs[c], s);
    atomicAdd(&cq[c], q);
  }
}
__global__ void gps_bnfinal(const float* __restrict__ cs, const float* __restrict__ cq,
                            const float* __restrict__ g, const float* __restrict__ beta,
                            float* __restrict__ sa, float* __restrict__ sb) {
  int c = threadIdx.x;
  const float invN = 1.0f / (float)NN;
  float m = cs[c] * invN;
  float v = cq[c] * invN - m * m;
  float a = g[c] * rsqrtf(v + BN_EPS);
  sa[c] = a;
  sb[c] = beta[c] - m * a;
}
__global__ void gps_bnapply_relu(const float* __restrict__ y, const float* __restrict__ sa,
                                 const float* __restrict__ sb, bf16_t* __restrict__ h, int n) {
  int i = blockIdx.x * blockDim.x + threadIdx.x;
  if (i < n) {
    int c = i & (HH - 1);
    float v = y[i] * sa[c] + sb[c];
    h[i] = f2bf(fmaxf(v, 0.0f));
  }
}

// ---------- launcher ---------------------------------------------------------
extern "C" void kernel_launch(void* const* d_in, const int* in_sizes, int n_in,
                              void* d_out, int out_size, void* d_ws, size_t ws_size,
                              hipStream_t stream) {
  (void)in_sizes; (void)n_in; (void)out_size; (void)ws_size;

  const float* x      = (const float*)d_in[0];
  const float* W_in   = (const float*)d_in[1];
  const float* b_in   = (const float*)d_in[2];
  const float* g_in   = (const float*)d_in[3];
  const float* beta_in= (const float*)d_in[4];
  const float* Wl     = (const float*)d_in[5];
  const float* bl     = (const float*)d_in[6];
  const float* Wr     = (const float*)d_in[7];
  const float* g_bn   = (const float*)d_in[8];
  const float* b_bn   = (const float*)d_in[9];
  const float* W_out  = (const float*)d_in[10];
  const float* b_out  = (const float*)d_in[11];
  const int*   edge   = (const int*)d_in[12];
  const int*   src    = edge;
  const int*   dst    = edge + EE;
  float* out          = (float*)d_out;

  // workspace carve-out (256B aligned slices)
  size_t off = 0;
  char* base = (char*)d_ws;
  auto take = [&](size_t bytes) -> void* {
    void* p = base + off;
    off += (bytes + 255) & ~(size_t)255;
    return p;
  };
  bf16_t* hB   = (bf16_t*)take((size_t)NN * HH * 2);   // node features, bf16
  bf16_t* aggB = (bf16_t*)take((size_t)NN * HH * 2);   // mean-aggregated, bf16
  float*  sbuf = (float*) take((size_t)NN * HH * 4);   // scatter accumulator
  float*  ybuf = (float*) take((size_t)NN * HH * 4);   // GEMM f32 output
  float*  invd = (float*) take((size_t)NN * 4);        // 1/degree
  bf16_t* WTin = (bf16_t*)take((size_t)HH * HH * 2);
  bf16_t* WlT  = (bf16_t*)take((size_t)LL * HH * HH * 2);
  bf16_t* WrT  = (bf16_t*)take((size_t)LL * HH * HH * 2);
  bf16_t* WoT  = (bf16_t*)take((size_t)48 * HH * 2);
  float*  cs   = (float*) take(HH * 4);
  float*  cq   = (float*) take(HH * 4);
  float*  sa   = (float*) take(HH * 4);
  float*  sb   = (float*) take(HH * 4);

  const int NH          = NN * HH;                       // 12.8M
  const int gNH         = (NH + 255) / 256;              // 50000
  const int gN          = (NN + 255) / 256;
  const int gE          = (EE + 255) / 256;
  const int tgrid       = (HH * HH + 255) / 256;
  const int gemm_blocks = (NN / 16 + 7) / 8;             // 8 waves / block
  const int bn_blocks   = (NN + 511) / 512;
  const size_t lds1     = (size_t)HH * HH * 2;           // 32KB
  const size_t lds2     = 2 * lds1;                      // 64KB
  const size_t ldsO     = (size_t)48 * HH * 2;           // 12KB

  // ---- prep: bf16 weights (pre-transposed), bf16 x, degrees ----
  gps_cvt_bf16<<<gNH, 256, 0, stream>>>(x, hB, NH);
  gps_transpose_w<<<tgrid, 256, 0, stream>>>(W_in, WTin, HH, HH, HH);
  for (int i = 0; i < LL; ++i) {
    gps_transpose_w<<<tgrid, 256, 0, stream>>>(Wl + (size_t)i * HH * HH, WlT + (size_t)i * HH * HH, HH, HH, HH);
    gps_transpose_w<<<tgrid, 256, 0, stream>>>(Wr + (size_t)i * HH * HH, WrT + (size_t)i * HH * HH, HH, HH, HH);
  }
  gps_transpose_w<<<(48 * HH + 255) / 256, 256, 0, stream>>>(W_out, WoT, HH, CC, 48);

  gps_zero_f32<<<gN, 256, 0, stream>>>(invd, NN);
  gps_degree<<<gE, 256, 0, stream>>>(dst, invd, EE);
  gps_invdeg<<<gN, 256, 0, stream>>>(invd, NN);

  // ---- input projection + BN + ReLU ----
  gps_gemm128<<<gemm_blocks, 256, lds1, stream>>>(hB, WTin, nullptr, nullptr, b_in, ybuf, NN);
  gps_zero_f32<<<1, 128, 0, stream>>>(cs, HH);
  gps_zero_f32<<<1, 128, 0, stream>>>(cq, HH);
  gps_bnstats<<<bn_blocks, 256, 0, stream>>>(ybuf, cs, cq, NN);
  gps_bnfinal<<<1, 128, 0, stream>>>(cs, cq, g_in, beta_in, sa, sb);
  gps_bnapply_relu<<<gNH, 256, 0, stream>>>(ybuf, sa, sb, hB, NH);

  // ---- SAGE layers ----
  for (int i = 0; i < LL; ++i) {
    gps_zero_f32x4<<<(NH / 4 + 255) / 256, 256, 0, stream>>>((v4f*)sbuf, NH / 4);
    gps_scatter<<<(EE + 31) / 32, 128, 0, stream>>>(hB, src, dst, sbuf, EE);
    gps_mean_bf16<<<gNH, 256, 0, stream>>>(sbuf, invd, aggB, NH);
    gps_gemm128<<<gemm_blocks, 256, lds2, stream>>>(
        aggB, WlT + (size_t)i * HH * HH, hB, WrT + (size_t)i * HH * HH,
        bl + (size_t)i * HH, ybuf, NN);
    if (i < LL - 1) {
      gps_zero_f32<<<1, 128, 0, stream>>>(cs, HH);
      gps_zero_f32<<<1, 128, 0, stream>>>(cq, HH);
      gps_bnstats<<<bn_blocks, 256, 0, stream>>>(ybuf, cs, cq, NN);
      gps_bnfinal<<<1, 128, 0, stream>>>(cs, cq, g_bn + (size_t)i * HH, b_bn + (size_t)i * HH, sa, sb);
      gps_bnapply_relu<<<gNH, 256, 0, stream>>>(ybuf, sa, sb, hB, NH);
    } else {
      gps_cvt_bf16<<<gNH, 256, 0, stream>>>(ybuf, hB, NH);
    }
  }

  // ---- output projection ----
  gps_gemm_out<<<gemm_blocks, 256, ldsO, stream>>>(hB, WoT, b_out, out, NN);
}
